// MultiheadAttention_73177652789921
// MI455X (gfx1250) — compile-verified, matrix operand order checked by reading, not run on hardware
//
#include <hip/hip_runtime.h>
#include <hip/hip_bf16.h>

// ---- problem constants (from reference) ----
#define BB 4
#define LL 2048
#define DD 512
#define HH 8
#define DHH 64   // D_HEAD
// TEMP = sqrt(64) = 8.  Softmax done in base-2 domain:
// fold (1/8)*log2(e) into Q so all exponentials are exp2f (bare v_exp_f32).
#define QSCALE 0.18033688011112042f   // 0.125 * log2(e)

typedef __attribute__((ext_vector_type(16))) __bf16          v16bf;
typedef __attribute__((ext_vector_type(8)))  float           v8f;
typedef __attribute__((ext_vector_type(4)))  float           v4f;
typedef __attribute__((ext_vector_type(16))) unsigned short  v16u16;
typedef __attribute__((ext_vector_type(8)))  unsigned short  v8u16;
typedef __attribute__((ext_vector_type(4)))  unsigned short  v4u16;

__device__ __forceinline__ unsigned short f2bf(float x) {
    unsigned u = __builtin_bit_cast(unsigned, x);
    u += 0x7FFFu + ((u >> 16) & 1u);        // round-to-nearest-even
    return (unsigned short)(u >> 16);
}

// Fragment loader for 16-bit A/B operands (wave32 layout, CDNA5 ISA 7.12.2):
// lane%16 = row; lanes 0-15 hold k {0..7,16..23}, lanes 16-31 hold k {8..15,24..31}.
// rowptr = this lane's row base; kofs = k0 + (lane>>4)*8.  Two b128 loads.
__device__ __forceinline__ v16bf frag_from_bf16(const unsigned short* rowptr, int kofs) {
    v8u16 lo = *(const v8u16*)(rowptr + kofs);
    v8u16 hi = *(const v8u16*)(rowptr + kofs + 16);
    v16u16 t;
#pragma unroll
    for (int i = 0; i < 8; ++i) { t[i] = lo[i]; t[i + 8] = hi[i]; }
    return __builtin_bit_cast(v16bf, t);
}

__device__ __forceinline__ v8f wmma_bf16(v16bf a, v16bf b, v8f c) {
    return __builtin_amdgcn_wmma_f32_16x16x32_bf16(
        /*neg_a=*/false, a, /*neg_b=*/false, b,
        /*c_mod=*/(short)0, c, /*reuse_a=*/false, /*reuse_b=*/false);
}

// ---------------------------------------------------------------------------
// Kernel 0: one-shot f32 -> bf16 conversion (vectorized x4).  Removes all
// per-tile redundant converts from the matmul inner loops.
// ---------------------------------------------------------------------------
__global__ void f32_to_bf16_kernel(const float* __restrict__ src,
                                   unsigned short* __restrict__ dst, int n4) {
    const int i = blockIdx.x * blockDim.x + threadIdx.x;
    if (i < n4) {
        v4f v = *(const v4f*)(src + (size_t)i * 4);
        v4u16 o;
#pragma unroll
        for (int k = 0; k < 4; ++k) o[k] = f2bf(v[k]);
        *(v4u16*)(dst + (size_t)i * 4) = o;
    }
}

// ---------------------------------------------------------------------------
// Kernel 1: QKV projection (all-bf16 operands).  One wave -> one
// 16(token) x 16(feature) tile, K=512 in 16 WMMA steps.
// grid = (128, 96), block = 128 (4 waves).  y: which=y/32, h=(y%32)/4, etile=y%4.
// ---------------------------------------------------------------------------
__global__ void qkv_proj_kernel(const unsigned short* __restrict__ qbf,
                                const unsigned short* __restrict__ wbf,
                                const float* __restrict__ bias,
                                unsigned short* __restrict__ Qu,
                                unsigned short* __restrict__ Ku,
                                unsigned short* __restrict__ Vu) {
    const int lane  = threadIdx.x & 31;
    const int wid   = threadIdx.x >> 5;
    const int tilem = blockIdx.x * 4 + wid;          // 0..511 token tiles
    const int ycol  = blockIdx.y;                    // 0..95
    const int which = ycol / 32;                     // 0=Q,1=K,2=V
    const int rem   = ycol % 32;
    const int h     = rem >> 2;
    const int etile = rem & 3;

    const int half = lane >> 4, lr = lane & 15;
    const int kl   = half * 8;
    const int e_l  = etile * 16 + lr;                 // feature col for this lane
    const int f    = h * (3 * DHH) + e_l * 3 + which; // row in in_weight

    const unsigned short* arow = qbf + (size_t)(tilem * 16 + lr) * DD;
    const unsigned short* brow = wbf + (size_t)f * DD;

    v8f c = {};
#pragma unroll
    for (int k0 = 0; k0 < DD; k0 += 32) {
        v16bf a = frag_from_bf16(arow, k0 + kl);
        v16bf b = frag_from_bf16(brow, k0 + kl);
        c = wmma_bf16(a, b, c);
    }
    const float bv = bias[f];
#pragma unroll
    for (int r = 0; r < 8; ++r) {
        const int t  = tilem * 16 + r + 8 * half;   // global token
        const int bb = t >> 11;                     // / LL
        const int l  = t & (LL - 1);
        const float v = c[r] + bv;
        if (which == 0) {
            Qu[(((size_t)bb * HH + h) * LL + l) * DHH + e_l] = f2bf(v * QSCALE);
        } else if (which == 1) {
            Ku[(((size_t)bb * HH + h) * LL + l) * DHH + e_l] = f2bf(v);
        } else {
            Vu[(size_t)t * DD + h * DHH + e_l] = f2bf(v);   // heads layout [b][l][h*64+e]
        }
    }
}

// ---------------------------------------------------------------------------
// Kernel 2a: per-(b,h,16-query-tile) online softmax stats (base-2 domain)
// over all 2048 keys.  One wave per tile; 128 j-tiles x 2 WMMA.
// waves = 4*8*128 = 4096.  Writes row-max M and reciprocal row-sum Zinv.
// ---------------------------------------------------------------------------
__global__ void softmax_stats_kernel(const unsigned short* __restrict__ Qu,
                                     const unsigned short* __restrict__ Ku,
                                     float* __restrict__ Mrow,
                                     float* __restrict__ Zinv) {
    const int lane = threadIdx.x & 31;
    const int wg   = blockIdx.x * (blockDim.x >> 5) + (threadIdx.x >> 5);
    const int itile = wg & 127;
    const int h     = (wg >> 7) & 7;
    const int b     = wg >> 10;
    const int half = lane >> 4, lr = lane & 15, kl = half * 8;

    const unsigned short* qrow =
        Qu + (((size_t)b * HH + h) * LL + itile * 16 + lr) * DHH;
    const unsigned short* kbase = Ku + ((size_t)b * HH + h) * LL * DHH;

    const v16bf a0 = frag_from_bf16(qrow, 0 + kl);
    const v16bf a1 = frag_from_bf16(qrow, 32 + kl);

    float m[8], z[8];
#pragma unroll
    for (int r = 0; r < 8; ++r) { m[r] = -3.0e38f; z[r] = 0.f; }

    for (int jt = 0; jt < LL / 16; ++jt) {
        const unsigned short* krow = kbase + (size_t)(jt * 16 + lr) * DHH;
        if (jt + 1 < LL / 16)                       // gfx1250 global_prefetch_b8
            __builtin_prefetch(krow + 16 * DHH, 0, 3);
        v16bf b0 = frag_from_bf16(krow, 0 + kl);
        v16bf b1 = frag_from_bf16(krow, 32 + kl);
        v8f c = {};
        c = wmma_bf16(a0, b0, c);
        c = wmma_bf16(a1, b1, c);
#pragma unroll
        for (int r = 0; r < 8; ++r) {
            const float s = c[r];
            if (s > m[r]) { z[r] = z[r] * exp2f(m[r] - s) + 1.0f; m[r] = s; }
            else          { z[r] += exp2f(s - m[r]); }
        }
    }
    // reduce (m,z) across the 16 lanes of each half (lane%16 spans keys)
#pragma unroll
    for (int off = 1; off < 16; off <<= 1) {
#pragma unroll
        for (int r = 0; r < 8; ++r) {
            const float m2 = __shfl_xor(m[r], off, 32);
            const float z2 = __shfl_xor(z[r], off, 32);
            const float mm = fmaxf(m[r], m2);
            z[r] = z[r] * exp2f(m[r] - mm) + z2 * exp2f(m2 - mm);
            m[r] = mm;
        }
    }
    if (lr == 0) {
#pragma unroll
        for (int r = 0; r < 8; ++r) {
            const int i = itile * 16 + r + 8 * half;
            Mrow[((size_t)b * HH + h) * LL + i] = m[r];
            Zinv[((size_t)b * HH + h) * LL + i] = __builtin_amdgcn_rcpf(z[r]);
        }
    }
}

// ---------------------------------------------------------------------------
// Kernel 2b: recompute scores per (b, i-tile, j-tile), sum softmax probs over
// the 8 heads, zero where mask, write masks output.  waves = 4*128*128 = 65536.
// ---------------------------------------------------------------------------
__global__ void attn_masks_kernel(const unsigned short* __restrict__ Qu,
                                  const unsigned short* __restrict__ Ku,
                                  const float* __restrict__ Mrow,
                                  const float* __restrict__ Zinv,
                                  const unsigned char* __restrict__ mask,
                                  float* __restrict__ masks_out) {
    const int lane = threadIdx.x & 31;
    const int wg   = blockIdx.x * (blockDim.x >> 5) + (threadIdx.x >> 5);
    const int jt = wg & 127;
    const int it = (wg >> 7) & 127;
    const int b  = wg >> 14;
    const int half = lane >> 4, lr = lane & 15, kl = half * 8;

    float acc[8];
#pragma unroll
    for (int r = 0; r < 8; ++r) acc[r] = 0.f;

    for (int h = 0; h < HH; ++h) {
        const unsigned short* qrow =
            Qu + (((size_t)b * HH + h) * LL + it * 16 + lr) * DHH;
        const unsigned short* krow =
            Ku + (((size_t)b * HH + h) * LL + jt * 16 + lr) * DHH;
        v16bf a0 = frag_from_bf16(qrow, 0 + kl);
        v16bf a1 = frag_from_bf16(qrow, 32 + kl);
        v16bf b0 = frag_from_bf16(krow, 0 + kl);
        v16bf b1 = frag_from_bf16(krow, 32 + kl);
        v8f c = {};
        c = wmma_bf16(a0, b0, c);
        c = wmma_bf16(a1, b1, c);
        const float* Mh = Mrow + ((size_t)b * HH + h) * LL;
        const float* Zh = Zinv + ((size_t)b * HH + h) * LL;
#pragma unroll
        for (int r = 0; r < 8; ++r) {
            const int i = it * 16 + r + 8 * half;
            acc[r] += exp2f(c[r] - Mh[i]) * Zh[i];
        }
    }
    const int j = jt * 16 + lr;
#pragma unroll
    for (int r = 0; r < 8; ++r) {
        const int i = it * 16 + r + 8 * half;
        const size_t idx = ((size_t)b * LL + i) * LL + j;
        masks_out[idx] = mask[idx] ? 0.f : acc[r];
    }
}

// ---------------------------------------------------------------------------
// Kernel 3: output projection: outputs = heads(=V) @ out_w^T + out_b.
// waves = 512*32 = 16384, 16 WMMA steps each (all-bf16 operands).
// ---------------------------------------------------------------------------
__global__ void out_proj_kernel(const unsigned short* __restrict__ Vu,
                                const unsigned short* __restrict__ owbf,
                                const float* __restrict__ out_b,
                                float* __restrict__ out) {
    const int lane = threadIdx.x & 31;
    const int wg   = blockIdx.x * (blockDim.x >> 5) + (threadIdx.x >> 5);
    const int tilen = wg & 31;
    const int tilem = wg >> 5;
    const int half = lane >> 4, lr = lane & 15, kl = half * 8;

    const unsigned short* arow = Vu + (size_t)(tilem * 16 + lr) * DD;
    const int o_l = tilen * 16 + lr;
    const unsigned short* brow = owbf + (size_t)o_l * DD;

    v8f c = {};
#pragma unroll
    for (int k0 = 0; k0 < DD; k0 += 32) {
        v16bf a = frag_from_bf16(arow, k0 + kl);
        v16bf b = frag_from_bf16(brow, k0 + kl);
        c = wmma_bf16(a, b, c);
    }
    const float bv = out_b[o_l];
#pragma unroll
    for (int r = 0; r < 8; ++r) {
        const int t = tilem * 16 + r + 8 * half;
        out[(size_t)t * DD + o_l] = c[r] + bv;
    }
}

// ---------------------------------------------------------------------------
extern "C" void kernel_launch(void* const* d_in, const int* in_sizes, int n_in,
                              void* d_out, int out_size, void* d_ws, size_t ws_size,
                              hipStream_t stream) {
    (void)in_sizes; (void)n_in; (void)out_size; (void)ws_size;

    const float*         q         = (const float*)d_in[0];
    const unsigned char* mask      = (const unsigned char*)d_in[1]; // bool array
    const float*         in_weight = (const float*)d_in[2];
    const float*         in_bias   = (const float*)d_in[3];
    const float*         out_w     = (const float*)d_in[4];
    const float*         out_b     = (const float*)d_in[5];

    float* out       = (float*)d_out;                    // [B,L,D] outputs
    float* masks_out = out + (size_t)BB * LL * DD;       // [B,L,L] masks

    // workspace layout (bytes):
    //  Qu 8MiB | Ku 8MiB | Vu 8MiB | M 256KiB | Zinv 256KiB |
    //  qbf 8MiB | wbf 1.5MiB | owbf 0.5MiB          (total ~35 MiB, L2-resident)
    char* ws = (char*)d_ws;
    unsigned short* Qu   = (unsigned short*)(ws);
    unsigned short* Ku   = (unsigned short*)(ws + (size_t)8  * 1024 * 1024);
    unsigned short* Vu   = (unsigned short*)(ws + (size_t)16 * 1024 * 1024);
    float*          Mr   = (float*)(ws + (size_t)24 * 1024 * 1024);
    float*          Zi   = (float*)(ws + (size_t)24 * 1024 * 1024 + 256 * 1024);
    unsigned short* qbf  = (unsigned short*)(ws + (size_t)25 * 1024 * 1024);
    unsigned short* wbf  = (unsigned short*)(ws + (size_t)33 * 1024 * 1024);
    unsigned short* owbf = (unsigned short*)(ws + (size_t)34 * 1024 * 1024 + 512 * 1024);

    const dim3 blk(128);  // 4 wave32 waves per workgroup

    // one-shot f32 -> bf16 conversions (counts are multiples of 4*256)
    f32_to_bf16_kernel<<<dim3(4096), dim3(256), 0, stream>>>(q, qbf, (BB * LL * DD) / 4);
    f32_to_bf16_kernel<<<dim3(768),  dim3(256), 0, stream>>>(in_weight, wbf, (3 * DD * DD) / 4);
    f32_to_bf16_kernel<<<dim3(256),  dim3(256), 0, stream>>>(out_w, owbf, (DD * DD) / 4);

    qkv_proj_kernel<<<dim3(128, 96), blk, 0, stream>>>(qbf, wbf, in_bias, Qu, Ku, Vu);
    softmax_stats_kernel<<<dim3(1024), blk, 0, stream>>>(Qu, Ku, Mr, Zi);
    attn_masks_kernel<<<dim3(16384), blk, 0, stream>>>(Qu, Ku, Mr, Zi, mask, masks_out);
    out_proj_kernel<<<dim3(4096), blk, 0, stream>>>(Vu, owbf, out_b, out);
}
